// SemanticDiscriminator_73967926772457
// MI455X (gfx1250) — matrix operand order repaired and checked by reading, not compiled, not run on hardware
//
#include <hip/hip_runtime.h>
#include <hip/hip_bf16.h>

// ---------------------------------------------------------------------------
// SemanticDiscriminator on MI455X (gfx1250, wave32, WMMA)
//   pool -> embed -> [xw GEMM + LSTM recurrence] x4 -> MLP head
// Input projections hoisted off the serial path into wide WMMA GEMMs
// (register-blocked 4 N-tiles/wave); recurrence keeps only K=512 h@Wh on
// tensor cores. h in LDS, c in VGPRs, next-step xw lines prefetched.
// ---------------------------------------------------------------------------

typedef __attribute__((ext_vector_type(16))) _Float16 v16h;
typedef __attribute__((ext_vector_type(8)))  _Float16 v8h;
typedef __attribute__((ext_vector_type(8)))  float    v8f;

static constexpr int BATCH = 256;
static constexpr int TSEQ  = 64;
static constexpr int EMB   = 256;
static constexpr int UNITS = 512;

__device__ __forceinline__ v8f zero8() {
  v8f z;
#pragma unroll
  for (int e = 0; e < 8; ++e) z[e] = 0.0f;
  return z;
}

__device__ __forceinline__ float sigmoidf_(float x) {
  return 1.0f / (1.0f + __expf(-x));
}

// A-fragment (16x32 f16, MxK): lane holds row m = lane&15.
// hi = lane>>4 selects K range [hi*16, hi*16+16).
__device__ __forceinline__ v16h load_a_frag(const _Float16* row32, int hi) {
  v8h lo = *(const v8h*)(row32 + hi * 16);
  v8h hv = *(const v8h*)(row32 + hi * 16 + 8);
  v16h a;
#pragma unroll
  for (int e = 0; e < 8; ++e) { a[e] = lo[e]; a[8 + e] = hv[e]; }
  return a;
}

// ---------------------------------------------------------------------------
// Elementwise prep kernels
// ---------------------------------------------------------------------------

__global__ void pool_kernel(const float* __restrict__ enc, float* __restrict__ feats) {
  int idx = blockIdx.x * blockDim.x + threadIdx.x;   // [0, 256*2048)
  int b = idx >> 11, ch = idx & 2047;
  const float* p = enc + (size_t)b * 100 * 2048 + ch;
  float s = 0.0f;
#pragma unroll 4
  for (int i = 0; i < 100; ++i) s += p[(size_t)i * 2048];
  feats[(size_t)b * 3072 + ch] = s * 0.01f;
}

__global__ void embed_kernel(const int* __restrict__ seq, const float* __restrict__ emb,
                             _Float16* __restrict__ tok) {
  int idx = blockIdx.x * blockDim.x + threadIdx.x;   // [0, 16384*256)
  int row = idx >> 8, col = idx & 255;
  int s = seq[row];
  tok[idx] = (_Float16)emb[(size_t)s * EMB + col];
}

// out[c*ostride + koff + r] = (f16) in[r*C + c]   (build n-major Wt)
__global__ void transcvt_kernel(const float* __restrict__ in, _Float16* __restrict__ out,
                                int R, int C, int ostride, int koff) {
  int idx = blockIdx.x * blockDim.x + threadIdx.x;
  if (idx >= R * C) return;
  int r = idx / C, c = idx - r * C;
  out[(size_t)c * ostride + koff + r] = (_Float16)in[idx];
}

__global__ void cvt_f32_f16(const float* __restrict__ in, _Float16* __restrict__ out, int cnt) {
  int idx = blockIdx.x * blockDim.x + threadIdx.x;
  if (idx < cnt) out[idx] = (_Float16)in[idx];
}

// ---------------------------------------------------------------------------
// WMMA GEMM, register-blocked: each wave computes a 16x64 output strip
// (4 N-tiles), reusing one A-fragment across 4 WMMAs per K-step.
// grid (M/16, N/512), block 256 = 8 waves.
// C[M,N] = act(A[M,K] @ Bt[N,K]^T + bias), f32 and/or f16 out.
// ---------------------------------------------------------------------------
__global__ __launch_bounds__(256)
void gemm_kernel(const _Float16* __restrict__ A, const _Float16* __restrict__ Bt,
                 const float* __restrict__ bias, float* __restrict__ Cf,
                 _Float16* __restrict__ Ch, int M, int N, int K, int relu)
{
  const int tid = threadIdx.x, w = tid >> 5, lane = tid & 31;
  const int n = lane & 15, hi = lane >> 4, hi8 = hi * 8;
  const int m0 = blockIdx.x * 16;
  const int n0 = (blockIdx.y * 8 + w) * 64;                 // 4 consecutive N-tiles

  v8f acc[4];
#pragma unroll
  for (int v = 0; v < 4; ++v) acc[v] = zero8();

  const _Float16* arow = A + (size_t)(m0 + n) * K;
  const _Float16* brow[4];
#pragma unroll
  for (int v = 0; v < 4; ++v)
    brow[v] = Bt + (size_t)(n0 + v * 16 + n) * K + hi * 16;

#pragma unroll 2
  for (int kb = 0; kb < K; kb += 32) {
    __builtin_prefetch(brow[0] + kb + 256, 0, 1);           // global_prefetch_b8
    v16h af = load_a_frag(arow + kb, hi);
    v16h bf[4];
#pragma unroll
    for (int v = 0; v < 4; ++v) bf[v] = *(const v16h*)(brow[v] + kb);
#pragma unroll
    for (int v = 0; v < 4; ++v)
      acc[v] = __builtin_amdgcn_wmma_f32_16x16x32_f16(
          false, af, false, bf[v], (short)0, acc[v], false, false);
  }

#pragma unroll
  for (int v = 0; v < 4; ++v) {
    const float bv = bias ? bias[n0 + v * 16 + n] : 0.0f;
#pragma unroll
    for (int r = 0; r < 8; ++r) {
      float val = acc[v][r] + bv;
      if (relu) val = fmaxf(val, 0.0f);
      size_t o = (size_t)(m0 + r + hi8) * N + n0 + v * 16 + n;
      if (Cf) Cf[o] = val;
      if (Ch) Ch[o] = (_Float16)val;
    }
  }
}

// ---------------------------------------------------------------------------
// LSTM recurrence:  z_t = xw[:,t,:] + h_{t-1} @ Wh,  Keras gates i,f,c,o.
// xw (= x@Wi + b) precomputed by gemm_kernel. grid.x = B/16 batch tiles;
// block = 512 threads = 16 waves; each wave owns 2 unit-tiles x 4 gates.
// Per K-step: all 8 B-fragments loaded first (one clause), then 8 WMMAs.
// Next timestep's xw cachelines are prefetched a full K-loop ahead.
// ---------------------------------------------------------------------------
__global__ __launch_bounds__(512)
void lstm_kernel(const _Float16* __restrict__ xw,   // [B*T, 2048] f16 (incl. bias)
                 const _Float16* __restrict__ wht,  // [2048, 512] n-major f16
                 const int*      __restrict__ seq,  // [B, T]
                 _Float16* __restrict__ yseq, int ycol0,              // [B*T,1024] or null
                 float*    __restrict__ yfin, int fstride, int fcol0, // or null
                 int reverse)
{
  __shared__ _Float16 hbuf[16 * UNITS];                               // 16 KB

  const int tid  = threadIdx.x;
  const int w    = tid >> 5;
  const int lane = tid & 31;
  const int n    = lane & 15;        // C/D column, A row, B column index
  const int hi   = lane >> 4;
  const int hi8  = hi * 8;
  const int b0   = blockIdx.x * 16;
  const int pg   = (lane >> 3) & 3;  // lanes spread prefetches across the 4 gates

  for (int i = tid; i < 16 * UNITS; i += 512) hbuf[i] = (_Float16)0.0f;

  float c[2][8];
  int   ucol[2];
  const _Float16* brow[2][4];
#pragma unroll
  for (int j = 0; j < 2; ++j) {
    ucol[j] = (w * 2 + j) * 16 + n;
#pragma unroll
    for (int r = 0; r < 8; ++r) c[j][r] = 0.0f;
#pragma unroll
    for (int g = 0; g < 4; ++g)
      brow[j][g] = wht + (size_t)(g * UNITS + ucol[j]) * UNITS + hi * 16;
  }
  __syncthreads();

  for (int s = 0; s < TSEQ; ++s) {
    const int t = reverse ? (TSEQ - 1 - s) : s;

    // Warm next step's xw lines (L2 -> near cache) while this step computes.
    const int tn = reverse ? (t > 0 ? t - 1 : 0) : (t < TSEQ - 1 ? t + 1 : t);
#pragma unroll
    for (int r = 0; r < 8; ++r)
      __builtin_prefetch(xw + ((size_t)(b0 + r + hi8) * TSEQ + tn) * 2048 +
                             pg * UNITS + w * 32 + n, 0, 1);

    // Seed accumulators with precomputed input projection (+bias).
    v8f acc[2][4];
#pragma unroll
    for (int r = 0; r < 8; ++r) {
      const _Float16* xrow = xw + ((size_t)(b0 + r + hi8) * TSEQ + t) * 2048;
#pragma unroll
      for (int j = 0; j < 2; ++j)
#pragma unroll
        for (int g = 0; g < 4; ++g)
          acc[j][g][r] = (float)xrow[g * UNITS + ucol[j]];
    }

    // Recurrent GEMM: acc += h @ Wh   (K = 512, 16 steps of 32)
    for (int kk = 0; kk < UNITS / 32; ++kk) {
      const int kbase = kk << 5;
      v16h afrag = load_a_frag(&hbuf[n * UNITS + kbase], hi);       // LDS
      v16h bf[2][4];                                                // batch loads
#pragma unroll
      for (int j = 0; j < 2; ++j)
#pragma unroll
        for (int g = 0; g < 4; ++g)
          bf[j][g] = *(const v16h*)(brow[j][g] + kbase);
#pragma unroll
      for (int j = 0; j < 2; ++j)
#pragma unroll
        for (int g = 0; g < 4; ++g)
          acc[j][g] = __builtin_amdgcn_wmma_f32_16x16x32_f16(
              false, afrag, false, bf[j][g], (short)0, acc[j][g], false, false);
    }
    __syncthreads();   // all LDS h reads done before updates

    bool msk[8];
#pragma unroll
    for (int r = 0; r < 8; ++r) msk[r] = seq[(b0 + r + hi8) * TSEQ + t] != 0;

#pragma unroll
    for (int j = 0; j < 2; ++j) {
#pragma unroll
      for (int r = 0; r < 8; ++r) {
        float iv = sigmoidf_(acc[j][0][r]);
        float fv = sigmoidf_(acc[j][1][r]);
        float gv = tanhf    (acc[j][2][r]);
        float ov = sigmoidf_(acc[j][3][r]);
        float cn = fv * c[j][r] + iv * gv;
        float hn = ov * tanhf(cn);
        int   mi = (r + hi8) * UNITS + ucol[j];
        float hout;
        if (msk[r]) { c[j][r] = cn; hout = hn; }
        else        { hout = (float)hbuf[mi]; }           // masked: carry state
        hbuf[mi] = (_Float16)hout;
        if (yseq)
          yseq[((size_t)(b0 + r + hi8) * TSEQ + t) * 1024 + ycol0 + ucol[j]] = (_Float16)hout;
      }
    }
    __syncthreads();   // h updated before next step's reads
  }

  if (yfin) {
#pragma unroll
    for (int j = 0; j < 2; ++j)
#pragma unroll
      for (int r = 0; r < 8; ++r)
        yfin[(size_t)(b0 + r + hi8) * fstride + fcol0 + ucol[j]] =
            (float)hbuf[(r + hi8) * UNITS + ucol[j]];
  }
}

__global__ void head_kernel(const float* __restrict__ h2, const float* __restrict__ W3,
                            const float* __restrict__ b3, float* __restrict__ out) {
  int b = threadIdx.x;                                      // 256 threads
  float s = 0.0f;
#pragma unroll 8
  for (int k = 0; k < 512; ++k) s += h2[(size_t)b * 512 + k] * W3[k];
  out[b] = 1.0f / (1.0f + __expf(-(s + b3[0])));
}

// ---------------------------------------------------------------------------
// Workspace layout (256B aligned), ~141 MB. One shared xw buffer is reused
// by all four (layer, direction) passes since the stream serializes them.
// ---------------------------------------------------------------------------
constexpr size_t al256(size_t x) { return (x + 255) & ~(size_t)255; }
constexpr size_t OFF_TOK   = 0;                                       // [16384,256]  f16
constexpr size_t OFF_WI1F  = OFF_TOK   + al256(16384ull * 256 * 2);   // [2048,256]   f16
constexpr size_t OFF_WI1B  = OFF_WI1F  + al256(2048ull * 256 * 2);
constexpr size_t OFF_WI2F  = OFF_WI1B  + al256(2048ull * 256 * 2);    // [2048,1024]  f16
constexpr size_t OFF_WI2B  = OFF_WI2F  + al256(2048ull * 1024 * 2);
constexpr size_t OFF_WH1F  = OFF_WI2B  + al256(2048ull * 1024 * 2);   // [2048,512]   f16
constexpr size_t OFF_WH1B  = OFF_WH1F  + al256(2048ull * 512 * 2);
constexpr size_t OFF_WH2F  = OFF_WH1B  + al256(2048ull * 512 * 2);
constexpr size_t OFF_WH2B  = OFF_WH2F  + al256(2048ull * 512 * 2);
constexpr size_t OFF_XW    = OFF_WH2B  + al256(2048ull * 512 * 2);    // [16384,2048] f16
constexpr size_t OFF_Y1    = OFF_XW    + al256(16384ull * 2048 * 2);  // [16384,1024] f16
constexpr size_t OFF_FEATS = OFF_Y1    + al256(16384ull * 1024 * 2);  // [256,3072]   f32
constexpr size_t OFF_FEATH = OFF_FEATS + al256(256ull * 3072 * 4);    // [256,3072]   f16
constexpr size_t OFF_W1T   = OFF_FEATH + al256(256ull * 3072 * 2);    // [1024,3072]  f16
constexpr size_t OFF_W2T   = OFF_W1T   + al256(3072ull * 1024 * 2);   // [512,1024]   f16
constexpr size_t OFF_H1H   = OFF_W2T   + al256(1024ull * 512 * 2);    // [256,1024]   f16
constexpr size_t OFF_H2F   = OFF_H1H   + al256(256ull * 1024 * 2);    // [256,512]    f32
constexpr size_t WS_NEED   = OFF_H2F   + al256(256ull * 512 * 4);

extern "C" void kernel_launch(void* const* d_in, const int* in_sizes, int n_in,
                              void* d_out, int out_size, void* d_ws, size_t ws_size,
                              hipStream_t stream) {
  (void)in_sizes; (void)n_in; (void)out_size;
  if (ws_size < WS_NEED) return;

  const float* enc    = (const float*)d_in[0];
  const int*   seq    = (const int*)d_in[1];
  const float* emb    = (const float*)d_in[2];
  const float* l1f_Wi = (const float*)d_in[3];
  const float* l1f_Wh = (const float*)d_in[4];
  const float* l1f_b  = (const float*)d_in[5];
  const float* l1b_Wi = (const float*)d_in[6];
  const float* l1b_Wh = (const float*)d_in[7];
  const float* l1b_b  = (const float*)d_in[8];
  const float* l2f_Wi = (const float*)d_in[9];
  const float* l2f_Wh = (const float*)d_in[10];
  const float* l2f_b  = (const float*)d_in[11];
  const float* l2b_Wi = (const float*)d_in[12];
  const float* l2b_Wh = (const float*)d_in[13];
  const float* l2b_b  = (const float*)d_in[14];
  const float* W1     = (const float*)d_in[15];
  const float* b1     = (const float*)d_in[16];
  const float* W2     = (const float*)d_in[17];
  const float* b2     = (const float*)d_in[18];
  const float* W3     = (const float*)d_in[19];
  const float* b3     = (const float*)d_in[20];
  float* out = (float*)d_out;

  char* ws = (char*)d_ws;
  _Float16* tok    = (_Float16*)(ws + OFF_TOK);
  _Float16* wi1ft  = (_Float16*)(ws + OFF_WI1F);
  _Float16* wi1bt  = (_Float16*)(ws + OFF_WI1B);
  _Float16* wi2ft  = (_Float16*)(ws + OFF_WI2F);
  _Float16* wi2bt  = (_Float16*)(ws + OFF_WI2B);
  _Float16* wh1ft  = (_Float16*)(ws + OFF_WH1F);
  _Float16* wh1bt  = (_Float16*)(ws + OFF_WH1B);
  _Float16* wh2ft  = (_Float16*)(ws + OFF_WH2F);
  _Float16* wh2bt  = (_Float16*)(ws + OFF_WH2B);
  _Float16* xwbuf  = (_Float16*)(ws + OFF_XW);
  _Float16* y1     = (_Float16*)(ws + OFF_Y1);
  float*    feats  = (float*)   (ws + OFF_FEATS);
  _Float16* featsh = (_Float16*)(ws + OFF_FEATH);
  _Float16* w1t    = (_Float16*)(ws + OFF_W1T);
  _Float16* w2t    = (_Float16*)(ws + OFF_W2T);
  _Float16* h1h    = (_Float16*)(ws + OFF_H1H);
  float*    h2f    = (float*)   (ws + OFF_H2F);

  // --- prep ---------------------------------------------------------------
  pool_kernel <<<(BATCH * 2048) / 256, 256, 0, stream>>>(enc, feats);
  embed_kernel<<<(BATCH * TSEQ * EMB) / 256, 256, 0, stream>>>(seq, emb, tok);

  auto tc = [&](const float* in, _Float16* o, int R, int C, int ostride, int koff) {
    int cnt = R * C;
    transcvt_kernel<<<(cnt + 255) / 256, 256, 0, stream>>>(in, o, R, C, ostride, koff);
  };
  tc(l1f_Wi, wi1ft,  256, 2048,  256, 0);
  tc(l1f_Wh, wh1ft,  512, 2048,  512, 0);
  tc(l1b_Wi, wi1bt,  256, 2048,  256, 0);
  tc(l1b_Wh, wh1bt,  512, 2048,  512, 0);
  tc(l2f_Wi, wi2ft, 1024, 2048, 1024, 0);
  tc(l2f_Wh, wh2ft,  512, 2048,  512, 0);
  tc(l2b_Wi, wi2bt, 1024, 2048, 1024, 0);
  tc(l2b_Wh, wh2bt,  512, 2048,  512, 0);
  tc(W1,     w1t,   3072, 1024, 3072, 0);
  tc(W2,     w2t,   1024,  512, 1024, 0);

  const dim3 gxw(16384 / 16, 2048 / 512);   // xw projection GEMMs: 4096 blocks

  // --- BiLSTM layer 1 (return_sequences -> y1) ----------------------------
  gemm_kernel<<<gxw, 256, 0, stream>>>(tok, wi1ft, l1f_b, nullptr, xwbuf, 16384, 2048, 256, 0);
  lstm_kernel<<<BATCH / 16, 512, 0, stream>>>(xwbuf, wh1ft, seq, y1,   0, nullptr, 0, 0, 0);
  gemm_kernel<<<gxw, 256, 0, stream>>>(tok, wi1bt, l1b_b, nullptr, xwbuf, 16384, 2048, 256, 0);
  lstm_kernel<<<BATCH / 16, 512, 0, stream>>>(xwbuf, wh1bt, seq, y1, 512, nullptr, 0, 0, 1);

  // --- BiLSTM layer 2 (final state -> feats[:, 2048:]) --------------------
  gemm_kernel<<<gxw, 256, 0, stream>>>(y1, wi2ft, l2f_b, nullptr, xwbuf, 16384, 2048, 1024, 0);
  lstm_kernel<<<BATCH / 16, 512, 0, stream>>>(xwbuf, wh2ft, seq, nullptr, 0, feats, 3072, 2048, 0);
  gemm_kernel<<<gxw, 256, 0, stream>>>(y1, wi2bt, l2b_b, nullptr, xwbuf, 16384, 2048, 1024, 0);
  lstm_kernel<<<BATCH / 16, 512, 0, stream>>>(xwbuf, wh2bt, seq, nullptr, 0, feats, 3072, 2560, 1);

  // --- MLP head -----------------------------------------------------------
  cvt_f32_f16<<<(BATCH * 3072) / 256, 256, 0, stream>>>(feats, featsh, BATCH * 3072);
  gemm_kernel<<<dim3(BATCH / 16, 1024 / 512), 256, 0, stream>>>(featsh, w1t, b1, nullptr, h1h, BATCH, 1024, 3072, 1);
  gemm_kernel<<<dim3(BATCH / 16,  512 / 512), 256, 0, stream>>>(h1h,    w2t, b2, h2f, nullptr, BATCH,  512, 1024, 1);
  head_kernel<<<1, BATCH, 0, stream>>>(h2f, W3, b3, out);
}
